// GCN_61186104099484
// MI455X (gfx1250) — compile-verified, matrix operand order checked by reading, not run on hardware
//
#include <hip/hip_runtime.h>
#include <hip/hip_bf16.h>

// ---------------------------------------------------------------------------
// 2-layer GCN: h = relu(Agg(x@W1)+b1); out = Agg(h@W2)+b2
// Agg(m)[i] = dinv[i]^2 * m[i]  +  sum_{e:(s->i)} dinv[s]*dinv[i]*m[s]
// GEMMs: bf16 operands pre-staged (x converted, W transposed) so every WMMA
// fragment is two contiguous b128 global loads (L2-resident); f32 accumulate
// via v_wmma_f32_16x16x32_bf16, register double-buffered, 32x64 wave tiles.
// ---------------------------------------------------------------------------

typedef __attribute__((ext_vector_type(16))) __bf16 v16bf;
typedef __attribute__((ext_vector_type(8)))  float  v8f;

union FragBF { v16bf v; uint4 q[2]; };

static constexpr int NN   = 50000;   // nodes
static constexpr int NE   = 400000;  // edges
static constexpr int DIN  = 512;
static constexpr int DHID = 512;
static constexpr int DOUT = 256;

// ---------------------------- degree / dinv --------------------------------

__global__ void deg_init_kernel(float* deg, int n) {
    int i = blockIdx.x * 256 + threadIdx.x;
    if (i < n) deg[i] = 1.0f;                       // self-loop contributes 1
}

__global__ void deg_count_kernel(const int* __restrict__ dst, float* deg, int ne) {
    int e = blockIdx.x * 256 + threadIdx.x;
    if (e < ne) atomicAdd(&deg[dst[e]], 1.0f);
}

__global__ void dinv_kernel(float* deg, int n) {
    int i = blockIdx.x * 256 + threadIdx.x;
    if (i < n) deg[i] = rsqrtf(deg[i]);             // deg >= 1 always
}

// ------------------------- operand staging to bf16 -------------------------

// elementwise f32 -> bf16, 8 elements/thread (b128 in, b128 out)
__global__ __launch_bounds__(256)
void cvt_bf16_kernel(const float* __restrict__ in, __bf16* __restrict__ out, int total8) {
    int idx = blockIdx.x * 256 + threadIdx.x;
    if (idx >= total8) return;
    const float4* p = (const float4*)in + (size_t)idx * 2;
    float4 v0 = p[0], v1 = p[1];
    union { __bf16 e[8]; uint4 u; } o;
    o.e[0] = (__bf16)v0.x; o.e[1] = (__bf16)v0.y; o.e[2] = (__bf16)v0.z; o.e[3] = (__bf16)v0.w;
    o.e[4] = (__bf16)v1.x; o.e[5] = (__bf16)v1.y; o.e[6] = (__bf16)v1.z; o.e[7] = (__bf16)v1.w;
    ((uint4*)out)[idx] = o.u;
}

// W[K][N] f32 -> WT[N][K] bf16 via 32x32 LDS tile (coalesced both sides)
__global__ __launch_bounds__(256)
void transpose_bf16_kernel(const float* __restrict__ W, __bf16* __restrict__ WT,
                           int K, int N) {
    __shared__ __bf16 tile[32][33];
    const int tid = threadIdx.x;
    const int nt = blockIdx.x * 32, kt = blockIdx.y * 32;
#pragma unroll
    for (int j = 0; j < 4; ++j) {
        int idx = j * 256 + tid, r = idx >> 5, c = idx & 31;
        tile[r][c] = (__bf16)W[(size_t)(kt + r) * N + (nt + c)];
    }
    __syncthreads();
#pragma unroll
    for (int j = 0; j < 4; ++j) {
        int idx = j * 256 + tid, r = idx >> 5, c = idx & 31;
        WT[(size_t)(nt + r) * K + (kt + c)] = tile[c][r];
    }
}

// ------------------------------ bf16 GEMM ----------------------------------
// C[M,N] = A[M,512] * B[512,N] with A bf16 row-major, B given as BT[N][512]
// bf16. Block: 256 thr = 8 waves; block tile 128x128; wave tile 32x64
// (2x4 WMMA 16x16 tiles -> 12 b128 loads per 8 WMMAs). Fragments register
// double-buffered over a fully unrolled K loop. No LDS, no barriers; the
// bf16 A matrix (51 MB) is L2-resident on MI455X (192 MB L2).

template <int N>
__global__ __launch_bounds__(256)
void gemm_bf16_kernel(const __bf16* __restrict__ A, const __bf16* __restrict__ BT,
                      float* __restrict__ C, int M) {
    constexpr int K = 512;
    const int tid  = threadIdx.x;
    const int lane = tid & 31;
    const int wave = tid >> 5;
    const int wm   = wave & 3;                  // 4 waves along M (32 rows each)
    const int wn   = wave >> 2;                 // 2 waves along N (64 cols each)
    const int m0   = blockIdx.y * 128;
    const int n0   = blockIdx.x * 128;
    const int lrow = lane & 15;
    const int g    = lane >> 4;                 // half-wave group -> K offset 8*g

    // fragment base pointers (A row clamped for M tail; stores stay guarded)
    const __bf16* Ap[2];
    const __bf16* Bp[4];
#pragma unroll
    for (int mi = 0; mi < 2; ++mi) {
        int row = m0 + wm * 32 + mi * 16 + lrow;
        if (row > M - 1) row = M - 1;
        Ap[mi] = A + (size_t)row * K + g * 8;
    }
#pragma unroll
    for (int ni = 0; ni < 4; ++ni) {
        int col = n0 + wn * 64 + ni * 16 + lrow;
        Bp[ni] = BT + (size_t)col * K + g * 8;
    }

    v8f acc[2][4] = {};
    FragBF fa[2][2], fb[2][4];                  // [parity][tile]

#define LOAD_FRAGS(P, K0)                                                   \
    do {                                                                    \
        _Pragma("unroll")                                                   \
        for (int t = 0; t < 2; ++t) {                                       \
            fa[P][t].q[0] = *(const uint4*)(Ap[t] + (K0));                  \
            fa[P][t].q[1] = *(const uint4*)(Ap[t] + (K0) + 16);             \
        }                                                                   \
        _Pragma("unroll")                                                   \
        for (int t = 0; t < 4; ++t) {                                       \
            fb[P][t].q[0] = *(const uint4*)(Bp[t] + (K0));                  \
            fb[P][t].q[1] = *(const uint4*)(Bp[t] + (K0) + 16);             \
        }                                                                   \
    } while (0)

    LOAD_FRAGS(0, 0);
#pragma unroll
    for (int kk = 0; kk < K / 32; ++kk) {
        const int cur = kk & 1, nxt = cur ^ 1;
        if (kk + 1 < K / 32) LOAD_FRAGS(nxt, (kk + 1) * 32);
#pragma unroll
        for (int mi = 0; mi < 2; ++mi)
#pragma unroll
            for (int ni = 0; ni < 4; ++ni)
                acc[mi][ni] = __builtin_amdgcn_wmma_f32_16x16x32_bf16(
                    false, fa[cur][mi].v, false, fb[cur][ni].v,
                    (short)0, acc[mi][ni], false, false);
    }
#undef LOAD_FRAGS

    // store: VGPR r holds M = r + 8*g, N = lane&15
#pragma unroll
    for (int mi = 0; mi < 2; ++mi)
#pragma unroll
        for (int ni = 0; ni < 4; ++ni) {
            int col = n0 + wn * 64 + ni * 16 + lrow;
#pragma unroll
            for (int r = 0; r < 8; ++r) {
                int row = m0 + wm * 32 + mi * 16 + g * 8 + r;
                if (row < M) C[(size_t)row * N + col] = acc[mi][ni][r];
            }
        }
}

// ------------------------------ aggregation --------------------------------

// out[i] = dinv[i]^2 * h[i]  (+ bias)  -- also fully initializes out
template <int D, bool BIAS>
__global__ __launch_bounds__(256)
void agg_init_kernel(const float* __restrict__ h, const float* __restrict__ dinv,
                     const float* __restrict__ bias, float* __restrict__ out, int n) {
    constexpr int CP = D / 4;
    int idx = blockIdx.x * 256 + threadIdx.x;
    if (idx >= n * CP) return;
    int i = idx / CP, c = idx % CP;
    float w = dinv[i]; w *= w;
    float4 v = ((const float4*)(h + (size_t)i * D))[c];
    float4 r = make_float4(v.x * w, v.y * w, v.z * w, v.w * w);
    if (BIAS) {
        float4 bv = ((const float4*)bias)[c];
        r.x += bv.x; r.y += bv.y; r.z += bv.z; r.w += bv.w;
    }
    ((float4*)(out + (size_t)i * D))[c] = r;
}

// out[dst[e]] += dinv[src]*dinv[dst] * h[src[e]]   (per-edge, per-float4-chunk)
template <int D>
__global__ __launch_bounds__(256)
void agg_edge_kernel(const float* __restrict__ h, const float* __restrict__ dinv,
                     const int* __restrict__ src, const int* __restrict__ dst,
                     float* out, int ne) {
    constexpr int CP = D / 4;
    int idx = blockIdx.x * 256 + threadIdx.x;
    if (idx >= ne * CP) return;
    int e = idx / CP, c = idx % CP;
    int s = src[e], d = dst[e];
    float w = dinv[s] * dinv[d];
    float4 v = ((const float4*)(h + (size_t)s * D))[c];
    float* o = out + (size_t)d * D + c * 4;
    atomicAdd(o + 0, v.x * w);
    atomicAdd(o + 1, v.y * w);
    atomicAdd(o + 2, v.z * w);
    atomicAdd(o + 3, v.w * w);
}

// z = relu(agg + b) -> bf16 (feeds second GEMM)
template <int D>
__global__ __launch_bounds__(256)
void bias_relu_bf16_kernel(const float* __restrict__ agg, const float* __restrict__ b,
                           __bf16* __restrict__ z, int n) {
    constexpr int CP = D / 4;
    int idx = blockIdx.x * 256 + threadIdx.x;
    if (idx >= n * CP) return;
    int c = idx % CP;
    float4 v = ((const float4*)agg)[idx];
    float4 bv = ((const float4*)b)[c];
    union { __bf16 e[4]; uint2 u; } o;
    o.e[0] = (__bf16)fmaxf(v.x + bv.x, 0.f);
    o.e[1] = (__bf16)fmaxf(v.y + bv.y, 0.f);
    o.e[2] = (__bf16)fmaxf(v.z + bv.z, 0.f);
    o.e[3] = (__bf16)fmaxf(v.w + bv.w, 0.f);
    ((uint2*)z)[idx] = o.u;
}

// ------------------------------- launcher ----------------------------------

extern "C" void kernel_launch(void* const* d_in, const int* in_sizes, int n_in,
                              void* d_out, int out_size, void* d_ws, size_t ws_size,
                              hipStream_t stream) {
    const float* x  = (const float*)d_in[0];
    const int*   ei = (const int*)d_in[1];      // [2, NE]: row 0 = src, row 1 = dst
    const float* W1 = (const float*)d_in[2];
    const float* b1 = (const float*)d_in[3];
    const float* W2 = (const float*)d_in[4];
    const float* b2 = (const float*)d_in[5];
    float* outp = (float*)d_out;
    const int* src = ei;
    const int* dst = ei + NE;

    // workspace layout (all offsets 256B-aligned)
    char* ws = (char*)d_ws;
    float*  dinv = (float*)(ws);                                  //  200 KB
    __bf16* xb   = (__bf16*)(ws + (size_t)262144);                // 51.2 MB (x_bf16, later z)
    __bf16* WT1  = (__bf16*)(ws + (size_t)51462144);              //  512 KB
    __bf16* WT2  = (__bf16*)(ws + (size_t)51986432);              //  256 KB
    float*  hbuf = (float*)(ws + (size_t)52248576);               // 102.4 MB (h, later h2)
    float*  abuf = (float*)(ws + (size_t)154648576);              // 102.4 MB (agg)

    const dim3 blk(256);

    // 1) symmetric normalization
    deg_init_kernel<<<dim3((NN + 255) / 256), blk, 0, stream>>>(dinv, NN);
    deg_count_kernel<<<dim3((NE + 255) / 256), blk, 0, stream>>>(dst, dinv, NE);
    dinv_kernel<<<dim3((NN + 255) / 256), blk, 0, stream>>>(dinv, NN);

    // 2) stage operands to bf16: x -> xb ; W1,W2 -> transposed bf16
    cvt_bf16_kernel<<<dim3((NN * DIN / 8 + 255) / 256), blk, 0, stream>>>(x, xb, NN * DIN / 8);
    transpose_bf16_kernel<<<dim3(DHID / 32, DIN / 32), blk, 0, stream>>>(W1, WT1, DIN, DHID);
    transpose_bf16_kernel<<<dim3(DOUT / 32, DHID / 32), blk, 0, stream>>>(W2, WT2, DHID, DOUT);

    // 3) h = x @ W1   (50000x512 @ 512x512)
    gemm_bf16_kernel<DHID><<<dim3(DHID / 128, (NN + 127) / 128), blk, 0, stream>>>(xb, WT1, hbuf, NN);

    // 4) agg1 = self-loop init + edge scatter
    agg_init_kernel<DHID, false><<<dim3((NN * (DHID / 4) + 255) / 256), blk, 0, stream>>>(
        hbuf, dinv, nullptr, abuf, NN);
    agg_edge_kernel<DHID><<<dim3((NE * (DHID / 4) + 255) / 256), blk, 0, stream>>>(
        hbuf, dinv, src, dst, abuf, NE);

    // 5) z = relu(agg1 + b1) -> bf16, reusing xb
    bias_relu_bf16_kernel<DHID><<<dim3((NN * (DHID / 4) + 255) / 256), blk, 0, stream>>>(
        abuf, b1, xb, NN);

    // 6) h2 = z @ W2  (50000x512 @ 512x256) -> reuse hbuf
    gemm_bf16_kernel<DOUT><<<dim3(DOUT / 128, (NN + 127) / 128), blk, 0, stream>>>(xb, WT2, hbuf, NN);

    // 7) out = self-loop init (+b2) + edge scatter
    agg_init_kernel<DOUT, true><<<dim3((NN * (DOUT / 4) + 255) / 256), blk, 0, stream>>>(
        hbuf, dinv, b2, outp, NN);
    agg_edge_kernel<DOUT><<<dim3((NE * (DOUT / 4) + 255) / 256), blk, 0, stream>>>(
        hbuf, dinv, src, dst, outp, NE);
}
